// LDABBVI_31164282700662
// MI455X (gfx1250) — matrix operand order repaired and checked by reading, not compiled, not run on hardware
//
#include <hip/hip_runtime.h>
#include <hip/hip_bf16.h>
#include <math.h>

#define N_TOPICS 64
#define VOCAB    32000
#define N_DOCS   2048
#define VT_PER_WAVE 10   // v-tiles per wave strip: 200 strips * 10 * 16 = 32000

typedef __attribute__((ext_vector_type(16))) __bf16 v16bf;
typedef __attribute__((ext_vector_type(8)))  __bf16 v8bf;
typedef __attribute__((ext_vector_type(8)))  float  v8f;

// ---------------- deterministic counter RNG ----------------
__device__ __forceinline__ unsigned long long splitmix64(unsigned long long x) {
    x += 0x9E3779B97F4A7C15ULL;
    x = (x ^ (x >> 30)) * 0xBF58476D1CE4E5B9ULL;
    x = (x ^ (x >> 27)) * 0x94D049BB133111EBULL;
    return x ^ (x >> 31);
}
__device__ __forceinline__ float u01(unsigned long long h) {
    return ((float)(unsigned int)(h >> 40) + 0.5f) * (1.0f / 16777216.0f);
}

// log of a Gamma(alpha) sample, alpha may be << 1.
// Marsaglia-Tsang for shape alpha+1 plus log-space boost: log X = log G(a+1) + log(U)/alpha.
// Fully deterministic in `base` so it can be re-evaluated in a second pass.
__device__ float sample_loggamma(unsigned long long base, float alpha) {
    float d = alpha + (2.0f / 3.0f);
    float c = __frsqrt_rn(9.0f * d);
    float res = __logf(d);          // fallback
    bool ok = false;
#pragma unroll 1
    for (int i = 0; i < 16 && !ok; ++i) {
        float u1 = u01(splitmix64(base + 3ULL * i));
        float u2 = u01(splitmix64(base + 3ULL * i + 1ULL));
        float u3 = u01(splitmix64(base + 3ULL * i + 2ULL));
        float z  = sqrtf(-2.0f * __logf(u1)) * __cosf(6.28318530717958647f * u2);
        float t  = 1.0f + c * z;
        if (t <= 0.0f) continue;
        float v3 = t * t * t;
        if (__logf(u3) < 0.5f * z * z + d - d * v3 + d * __logf(v3)) {
            res = __logf(d * v3);
            ok = true;
        }
    }
    float ub = u01(splitmix64(base + 63ULL));
    return res + __logf(ub) / alpha;   // log-space boost (alpha tiny -> large negative, LSE handles)
}

// ---------------- accumulators ----------------
// acc[0] = log_lik ; acc[1] = log q(topics) ; acc[2] = sum_{d,k} log doc_topics
__global__ void lda_init_kernel(float* acc) {
    if (threadIdx.x < 16) acc[threadIdx.x] = 0.0f;
}

// ---------------- topics: sample, normalize (LSE), emit bf16 topicsT[V][64] ----------------
__global__ __launch_bounds__(256) void lda_topics_kernel(
        const float* __restrict__ tlv,      // [K, V] topic_log_var
        __bf16* __restrict__ topicsT,       // [V, K] bf16
        float* __restrict__ acc)
{
    const int k   = blockIdx.x;             // topic
    const int tid = threadIdx.x;            // 0..255
    __shared__ float sm[256], ss[256];
    __shared__ float sh_lse, sh_lamsum, sh_lgam;

    // pass 1: online log-sum-exp over the row + lambda sums
    float m = -INFINITY, s = 0.0f, lamsum = 0.0f, lgsum = 0.0f;
    for (int v = tid; v < VOCAB; v += 256) {
        float lam = __expf(tlv[(size_t)k * VOCAB + v]);
        unsigned long long base = 0x1000000000000000ULL +
                                  ((unsigned long long)k * VOCAB + v) * 64ULL;
        float lx = sample_loggamma(base, lam);
        if (lx > m) { s = s * __expf(m - lx) + 1.0f; m = lx; }
        else        { s += __expf(lx - m); }
        lamsum += lam;
        lgsum  += lgammaf(lam);
    }
    sm[tid] = m; ss[tid] = s;
    __syncthreads();
    for (int w = 128; w > 0; w >>= 1) {
        if (tid < w) {
            float m1 = sm[tid], m2 = sm[tid + w];
            float M  = fmaxf(m1, m2);
            ss[tid]  = ss[tid] * __expf(m1 - M) + ss[tid + w] * __expf(m2 - M);
            sm[tid]  = M;
        }
        __syncthreads();
    }
    if (tid == 0) sh_lse = sm[0] + __logf(ss[0]);
    __syncthreads();
    const float lse = sh_lse;

    // reduce lambda sums
    sm[tid] = lamsum; ss[tid] = lgsum;
    __syncthreads();
    for (int w = 128; w > 0; w >>= 1) {
        if (tid < w) { sm[tid] += sm[tid + w]; ss[tid] += ss[tid + w]; }
        __syncthreads();
    }
    if (tid == 0) { sh_lamsum = sm[0]; sh_lgam = ss[0]; }
    __syncthreads();

    // pass 2: recompute (deterministic RNG), write bf16 topicsT, log q term
    float term = 0.0f;
    for (int v = tid; v < VOCAB; v += 256) {
        float lam = __expf(tlv[(size_t)k * VOCAB + v]);
        unsigned long long base = 0x1000000000000000ULL +
                                  ((unsigned long long)k * VOCAB + v) * 64ULL;
        float lt = sample_loggamma(base, lam) - lse;   // log topics[k][v]
        topicsT[(size_t)v * N_TOPICS + k] = (__bf16)__expf(lt);
        term += (lam - 1.0f) * lt;
    }
    sm[tid] = term;
    __syncthreads();
    for (int w = 128; w > 0; w >>= 1) {
        if (tid < w) sm[tid] += sm[tid + w];
        __syncthreads();
    }
    if (tid == 0)
        atomicAdd(&acc[1], sm[0] + lgammaf(sh_lamsum) - sh_lgam);
}

// ---------------- doc_topics: sample, normalize, emit bf16 dt[D][64] ----------------
__global__ __launch_bounds__(64) void lda_docs_kernel(
        __bf16* __restrict__ dtbf, float* __restrict__ acc)
{
    const int d = blockIdx.x;
    const int k = threadIdx.x;              // 0..63
    __shared__ float sm[64], ss[64];
    __shared__ float shM, shLse;

    const float alpha = 1.0f / (float)N_TOPICS;   // gam = exp(log(1/K))
    unsigned long long base = 0x2000000000000000ULL +
                              ((unsigned long long)d * N_TOPICS + k) * 64ULL;
    float lx = sample_loggamma(base, alpha);

    sm[k] = lx;
    __syncthreads();
    for (int w = 32; w > 0; w >>= 1) {
        if (k < w) sm[k] = fmaxf(sm[k], sm[k + w]);
        __syncthreads();
    }
    if (k == 0) shM = sm[0];
    __syncthreads();
    ss[k] = __expf(lx - shM);
    __syncthreads();
    for (int w = 32; w > 0; w >>= 1) {
        if (k < w) ss[k] += ss[k + w];
        __syncthreads();
    }
    if (k == 0) shLse = shM + __logf(ss[0]);
    __syncthreads();

    float lt = lx - shLse;                        // log doc_topics[d][k]
    dtbf[(size_t)d * N_TOPICS + k] = (__bf16)__expf(lt);

    sm[k] = lt;
    __syncthreads();
    for (int w = 32; w > 0; w >>= 1) {
        if (k < w) sm[k] += sm[k + w];
        __syncthreads();
    }
    if (k == 0) atomicAdd(&acc[2], sm[0]);
}

// ---------------- fused WMMA GEMM + masked log-likelihood ----------------
// One wave32 owns a strip of VT_PER_WAVE consecutive 16x16 tiles of word_probs.
// A operand (doc_topics 16x64) loaded once per wave, held in VGPRs across the strip.
// K=64 -> two chained V_WMMA_F32_16X16X32_BF16 per tile.
__global__ __launch_bounds__(256) void lda_gemm_loglik_kernel(
        const float*  __restrict__ docs,     // [D, V] f32
        const __bf16* __restrict__ topicsT,  // [V, K] bf16 (B operand, contiguous in K)
        const __bf16* __restrict__ dtbf,     // [D, K] bf16 (A operand, contiguous in K)
        float* __restrict__ acc)
{
    const int lane  = threadIdx.x & 31;
    const int wave  = threadIdx.x >> 5;
    const int d0    = blockIdx.y * 16;
    const int strip = blockIdx.x * 8 + wave;        // 0..199
    const int vbase = strip * (VT_PER_WAVE * 16);
    const int mn    = lane & 15;
    const int hi    = lane >> 4;             // lane half selects K-group / M-group

    // --- A operand: 16x32 bf16 (ISA layout: lanes0-15 K in {kb..kb+7, kb+16..kb+23}, kb = hi*8)
    const __bf16* arow = dtbf + (size_t)(d0 + mn) * N_TOPICS;
    const int kA = hi * 8;
    v8bf a0lo = *(const v8bf*)(arow + kA);
    v8bf a0hi = *(const v8bf*)(arow + kA + 16);
    v8bf a1lo = *(const v8bf*)(arow + 32 + kA);
    v8bf a1hi = *(const v8bf*)(arow + 32 + kA + 16);
    v16bf a0 = __builtin_shufflevector(a0lo, a0hi, 0,1,2,3,4,5,6,7,8,9,10,11,12,13,14,15);
    v16bf a1 = __builtin_shufflevector(a1lo, a1hi, 0,1,2,3,4,5,6,7,8,9,10,11,12,13,14,15);

    const int kB    = hi * 8;
    const int rbase = hi * 8;
    float part = 0.0f;

#pragma unroll 1
    for (int s = 0; s < VT_PER_WAVE; ++s) {
        const int v0 = vbase + s * 16;

        // prefetch next tile's docs row for this lane (global_prefetch_b8)
        if (s + 1 < VT_PER_WAVE)
            __builtin_prefetch(docs + (size_t)(d0 + rbase + (lane & 7)) * VOCAB + v0 + 16, 0, 0);

        // --- B operand: 32x16 bf16 (lanes0-15 hold K in {0..7,16..23}, lanes16-31 K in {8..15,24..31}; N = lane&15)
        const __bf16* brow = topicsT + (size_t)(v0 + mn) * N_TOPICS;
        v8bf b0lo = *(const v8bf*)(brow + kB);
        v8bf b0hi = *(const v8bf*)(brow + kB + 16);
        v8bf b1lo = *(const v8bf*)(brow + 32 + kB);
        v8bf b1hi = *(const v8bf*)(brow + 32 + kB + 16);
        v16bf b0 = __builtin_shufflevector(b0lo, b0hi, 0,1,2,3,4,5,6,7,8,9,10,11,12,13,14,15);
        v16bf b1 = __builtin_shufflevector(b1lo, b1hi, 0,1,2,3,4,5,6,7,8,9,10,11,12,13,14,15);

        v8f c = {};
        c = __builtin_amdgcn_wmma_f32_16x16x32_bf16(false, a0, false, b0, (short)0, c, false, false);
        c = __builtin_amdgcn_wmma_f32_16x16x32_bf16(false, a1, false, b1, (short)0, c, false, false);

        // --- branchless masked log-likelihood (C layout: VGPR r holds M = r + hi*8, N = lane&15)
        // docs ~80% nonzero => a wave essentially never skips a log; masking the weight
        // beats EXEC-branching around v_log_f32.
#pragma unroll
        for (int r = 0; r < 8; ++r) {
            float wp = c[r];
            float w  = docs[(size_t)(d0 + rbase + r) * VOCAB + v0 + mn];
            float lw = __logf(wp + 1e-10f);
            float wm = (w > 0.0f) ? w : 0.0f;
            part = fmaf(wm, lw, part);
        }
    }

    // wave reduce, then one atomic per block
#pragma unroll
    for (int off = 16; off > 0; off >>= 1)
        part += __shfl_down(part, off, 32);

    __shared__ float sacc[8];
    if (lane == 0) sacc[wave] = part;
    __syncthreads();
    if (threadIdx.x == 0) {
        float t = 0.0f;
#pragma unroll
        for (int i = 0; i < 8; ++i) t += sacc[i];
        atomicAdd(&acc[0], t);
    }
}

// ---------------- finalize: fold lgamma constants, emit -ELBO ----------------
__global__ void lda_finalize_kernel(const float* __restrict__ acc, float* __restrict__ out) {
    if (threadIdx.x != 0 || blockIdx.x != 0) return;
    const float log_lik   = acc[0];
    const float lq_topics = acc[1];
    const float sumlogdt  = acc[2];
    // priors are symmetric Dirichlet(1): (conc-1)=0, lgamma(1)=0
    const float log_p_topics = (float)N_TOPICS * lgammaf((float)VOCAB);
    const float log_p_doc    = (float)N_DOCS   * lgammaf((float)N_TOPICS);
    const float gam = 1.0f / (float)N_TOPICS;
    // log q(doc_topics): (gam-1)*sum log dt + D*(lgamma(K*gam) - K*lgamma(gam)), lgamma(1)=0
    const float lq_doc = (gam - 1.0f) * sumlogdt
                       - (float)N_DOCS * (float)N_TOPICS * lgammaf(gam);
    const float elbo = (log_p_topics + log_p_doc + log_lik) - (lq_topics + lq_doc);
    out[0] = -elbo;
}

// ---------------- launch ----------------
extern "C" void kernel_launch(void* const* d_in, const int* in_sizes, int n_in,
                              void* d_out, int out_size, void* d_ws, size_t ws_size,
                              hipStream_t stream) {
    (void)in_sizes; (void)n_in; (void)out_size; (void)ws_size;
    const float* docs = (const float*)d_in[0];            // [D, V]
    const float* tlv  = (const float*)d_in[1];            // [K, V]
    float* out = (float*)d_out;

    // workspace layout: 64B accumulators | topicsT bf16 [V][64] | dtbf bf16 [D][64]
    float*  acc     = (float*)d_ws;
    __bf16* topicsT = (__bf16*)((char*)d_ws + 64);
    __bf16* dtbf    = (__bf16*)((char*)d_ws + 64 + (size_t)VOCAB * N_TOPICS * 2);

    lda_init_kernel<<<1, 32, 0, stream>>>(acc);
    lda_topics_kernel<<<N_TOPICS, 256, 0, stream>>>(tlv, topicsT, acc);
    lda_docs_kernel<<<N_DOCS, 64, 0, stream>>>(dtbf, acc);
    dim3 grid(25, N_DOCS / 16);                           // 25 blocks * 8 waves = 200 strips
    lda_gemm_loglik_kernel<<<grid, 256, 0, stream>>>(docs, topicsT, dtbf, acc);
    lda_finalize_kernel<<<1, 32, 0, stream>>>(acc, out);
}
